// MMRecBlock_87497073754915
// MI455X (gfx1250) — matrix operand order repaired and checked by reading, not compiled, not run on hardware
//
#include <hip/hip_runtime.h>
#include <hip/hip_bf16.h>

// ---------------------------------------------------------------------------
// Types for CDNA5 WMMA (wave32)
// ---------------------------------------------------------------------------
typedef __attribute__((ext_vector_type(16))) __bf16 v16bf;
typedef __attribute__((ext_vector_type(8)))  float  v8f;

#define MODEL_DIM 2048
#define INNER_DIM 512
#define NUM_HEADS 16
#define HEAD_DIM  128
#define FFN_DIM   8192
#define MEM_TOKENS 64
#define BATCH 2
#define SEQ   2048
#define ROWS  (BATCH * SEQ)        // 4096
#define MROWS (BATCH * MEM_TOKENS) // 128

// ---------------------------------------------------------------------------
// helpers
// ---------------------------------------------------------------------------
__device__ inline __bf16 f2bf(float f) {
    unsigned u = __float_as_uint(f);
    unsigned r = (u + 0x7fffu + ((u >> 16) & 1u)) >> 16;   // round-nearest-even
    unsigned short h = (unsigned short)r;
    return *(__bf16*)&h;
}

__device__ inline float gelu_f(float x) {
    const float c = 0.7978845608028654f;  // sqrt(2/pi)
    float t = tanhf(c * (x + 0.044715f * x * x * x));
    return 0.5f * x * (1.f + t);
}

__device__ inline float sigmoid_f(float x) {
    return 1.f / (1.f + __expf(-x));
}

// ---------------------------------------------------------------------------
// WMMA GEMM with async-LDS double buffering.
//   C[M][N] = act(A[M][K](bf16) * Bt[N][K](bf16)^T + bias) (+resid)
// Block: 256 threads = 8 waves, wave grid 4(M) x 2(N); wave tile 32x64;
// block tile 128x128; K-step 64, staged in LDS via
// GLOBAL_LOAD_ASYNC_TO_LDS_B128 (ASYNCcnt) with double buffering.
// M, N multiples of 128; K multiple of 64.
// ---------------------------------------------------------------------------
#define KSTEP 64
#define LDS_ROWSTRIDE 144                 // 128B row + 16B pad (bank spread)
#define LDS_TILE (128 * LDS_ROWSTRIDE)    // 18432 B per A or B tile
#define LDS_BUF  (2 * LDS_TILE)           // A + B

// Issue one 128x64-bf16 A tile and one B tile into LDS (async, per-lane 16B).
__device__ inline void async_tile_load(unsigned ldsA, unsigned ldsB,
                                       const __bf16* __restrict__ gA,
                                       const __bf16* __restrict__ gB,
                                       int K, int tid) {
#pragma unroll
    for (int i = 0; i < 4; ++i) {
        const int c   = tid + (i << 8);   // 0..1023 : 16B chunk id
        const int row = c >> 3;           // 0..127
        const int ck  = c & 7;            // chunk within 128B row
        const unsigned la = ldsA + row * LDS_ROWSTRIDE + ck * 16;
        const unsigned lb = ldsB + row * LDS_ROWSTRIDE + ck * 16;
        const __bf16* ga = gA + (size_t)row * K + ck * 8;
        const __bf16* gb = gB + (size_t)row * K + ck * 8;
        asm volatile("global_load_async_to_lds_b128 %0, %1, off"
                     :: "v"(la), "v"((unsigned long long)(uintptr_t)ga)
                     : "memory");
        asm volatile("global_load_async_to_lds_b128 %0, %1, off"
                     :: "v"(lb), "v"((unsigned long long)(uintptr_t)gb)
                     : "memory");
    }
}

template<int ACT, int OMODE>
__device__ inline void store_tile(const v8f& acc, int row0, int n0, int lane,
                                  const float* __restrict__ bias,
                                  const float* __restrict__ resid,
                                  float* __restrict__ Cf,
                                  __bf16* __restrict__ Cb, int N) {
    const int n = n0 + (lane & 15);
    const int rbase = row0 + ((lane >> 4) << 3);
    const float bv = bias[n];
#pragma unroll
    for (int v = 0; v < 8; ++v) {
        float val = acc[v] + bv;
        if (ACT == 1) val = gelu_f(val);
        else if (ACT == 2) val = sigmoid_f(val);
        const size_t idx = (size_t)(rbase + v) * N + n;
        if (resid) val += resid[idx];
        if (OMODE & 1) Cf[idx] = val;
        if (OMODE & 2) Cb[idx] = f2bf(val);
    }
}

template<int ACT, int OMODE>
__global__ __launch_bounds__(256)
void gemm_bf16_wmma(const __bf16* __restrict__ A, const __bf16* __restrict__ Bt,
                    const float* __restrict__ bias, const float* __restrict__ resid,
                    float* __restrict__ Cf, __bf16* __restrict__ Cb,
                    int M, int N, int K) {
    __shared__ __align__(16) unsigned char smem[2 * LDS_BUF];   // 72 KB
    (void)M;

    const int tid  = threadIdx.x;
    const int lane = tid & 31;
    const int wid  = tid >> 5;
    const int wm   = wid & 3;    // 0..3 over M
    const int wn   = wid >> 2;   // 0..1 over N
    const int mBase = blockIdx.y * 128;
    const int nBase = blockIdx.x * 128;

    const unsigned ldsBase = (unsigned)(uintptr_t)(void*)smem;  // LDS offset

    v8f zero = {};
    v8f acc[2][4];
#pragma unroll
    for (int mt = 0; mt < 2; ++mt)
#pragma unroll
        for (int nt = 0; nt < 4; ++nt) acc[mt][nt] = zero;

    const __bf16* gA = A  + (size_t)mBase * K;
    const __bf16* gB = Bt + (size_t)nBase * K;

    const int mrow = lane & 15;
    const int hi   = lane >> 4;
    const int nTiles = K / KSTEP;

    // prologue: fill buffer 0
    async_tile_load(ldsBase, ldsBase + LDS_TILE, gA, gB, K, tid);
    asm volatile("s_wait_asynccnt 0x0" ::: "memory");
    __syncthreads();

    for (int kt = 0; kt < nTiles; ++kt) {
        // prefetch next K-tile into the other buffer (overlaps WMMA below)
        if (kt + 1 < nTiles) {
            const unsigned bufNxt = ldsBase + (unsigned)(((kt + 1) & 1) * LDS_BUF);
            async_tile_load(bufNxt, bufNxt + LDS_TILE,
                            gA + (size_t)(kt + 1) * KSTEP,
                            gB + (size_t)(kt + 1) * KSTEP, K, tid);
        }

        const unsigned char* smA = smem + (size_t)(kt & 1) * LDS_BUF;
        const unsigned char* smB = smA + LDS_TILE;

#pragma unroll
        for (int ks = 0; ks < 2; ++ks) {
            v16bf a[2];
            // A fragments (16-bit A 16x32 layout: lanes0-15 K{0..7,16..23},
            // lanes16-31 K{8..15,24..31})
#pragma unroll
            for (int mt = 0; mt < 2; ++mt) {
                const unsigned char* p = smA
                    + (size_t)(wm * 32 + mt * 16 + mrow) * LDS_ROWSTRIDE
                    + (size_t)(ks * 32 + hi * 8) * 2;
                ((uint4*)&a[mt])[0] = *(const uint4*)(p);
                ((uint4*)&a[mt])[1] = *(const uint4*)(p + 32);   // K + 16
            }
#pragma unroll
            for (int nt = 0; nt < 4; ++nt) {
                // B fragment (32x16): lanes0-15 K{0..7,16..23}, lanes16-31 K{8..15,24..31}
                const unsigned char* p = smB
                    + (size_t)(wn * 64 + nt * 16 + mrow) * LDS_ROWSTRIDE
                    + (size_t)(ks * 32 + hi * 8) * 2;
                v16bf b;
                ((uint4*)&b)[0] = *(const uint4*)(p);
                ((uint4*)&b)[1] = *(const uint4*)(p + 32);       // K + 16
                acc[0][nt] = __builtin_amdgcn_wmma_f32_16x16x32_bf16(
                    false, a[0], false, b, (short)0, acc[0][nt], false, false);
                acc[1][nt] = __builtin_amdgcn_wmma_f32_16x16x32_bf16(
                    false, a[1], false, b, (short)0, acc[1][nt], false, false);
            }
        }

        asm volatile("s_wait_asynccnt 0x0" ::: "memory");
        __syncthreads();
    }

#pragma unroll
    for (int mt = 0; mt < 2; ++mt)
#pragma unroll
        for (int nt = 0; nt < 4; ++nt)
            store_tile<ACT, OMODE>(acc[mt][nt],
                                   mBase + wm * 32 + mt * 16,
                                   nBase + wn * 64 + nt * 16,
                                   lane, bias, resid, Cf, Cb, N);
}

// ---------------------------------------------------------------------------
// Weight transpose + fp32 -> bf16 :  W[K][N] -> Wt[N][K]
// ---------------------------------------------------------------------------
__global__ __launch_bounds__(256)
void transpose_to_bf16(const float* __restrict__ W, __bf16* __restrict__ Wt,
                       int K, int N) {
    __shared__ float tile[32][33];
    const int kt = blockIdx.y * 32;
    const int nt = blockIdx.x * 32;
    for (int i = threadIdx.y; i < 32; i += 8)
        tile[i][threadIdx.x] = W[(size_t)(kt + i) * N + nt + threadIdx.x];
    __syncthreads();
    for (int i = threadIdx.y; i < 32; i += 8)
        Wt[(size_t)(nt + i) * K + kt + threadIdx.x] = f2bf(tile[threadIdx.x][i]);
}

// ---------------------------------------------------------------------------
// fp32 -> bf16 elementwise
// ---------------------------------------------------------------------------
__global__ void convert_bf16(const float* __restrict__ in, __bf16* __restrict__ out,
                             size_t n) {
    size_t i = (size_t)blockIdx.x * blockDim.x + threadIdx.x;
    if (i < n) out[i] = f2bf(in[i]);
}

// ---------------------------------------------------------------------------
// RMSNorm (per-row) -> bf16 :  out = w * x / (||x||/sqrt(D) + eps)
// ---------------------------------------------------------------------------
__global__ __launch_bounds__(256)
void rmsnorm_bf16(const float* __restrict__ x, const float* __restrict__ w,
                  __bf16* __restrict__ out, int D) {
    const int row = blockIdx.x;
    const float* xr = x + (size_t)row * D;
    __shared__ float red[256];
    float ss = 0.f;
    for (int i = threadIdx.x; i < D; i += 256) { float v = xr[i]; ss += v * v; }
    red[threadIdx.x] = ss;
    __syncthreads();
    for (int s = 128; s > 0; s >>= 1) {
        if (threadIdx.x < s) red[threadIdx.x] += red[threadIdx.x + s];
        __syncthreads();
    }
    const float norm  = sqrtf(red[0] / (float)D);
    const float scale = 1.f / (norm + 1e-6f);
    for (int i = threadIdx.x; i < D; i += 256)
        out[(size_t)row * D + i] = f2bf(w[i] * xr[i] * scale);
}

// ---------------------------------------------------------------------------
// gamma = sigmoid(a + b)
// ---------------------------------------------------------------------------
__global__ void sigmoid_add_kernel(const float* __restrict__ a,
                                   const float* __restrict__ b,
                                   float* __restrict__ out, size_t n) {
    size_t i = (size_t)blockIdx.x * blockDim.x + threadIdx.x;
    if (i < n) out[i] = sigmoid_f(a[i] + b[i]);
}

// ---------------------------------------------------------------------------
// h_t = z*gate + gamma*x  (fp32 + bf16 outputs)
// ---------------------------------------------------------------------------
__global__ void ht_kernel(const float* __restrict__ z, const float* __restrict__ gate,
                          const float* __restrict__ gamma, const float* __restrict__ x,
                          float* __restrict__ h, __bf16* __restrict__ h_bf, size_t n) {
    size_t i = (size_t)blockIdx.x * blockDim.x + threadIdx.x;
    if (i < n) {
        float v = z[i] * gate[i] + gamma[i] * x[i];
        h[i] = v;
        h_bf[i] = f2bf(v);
    }
}

// ---------------------------------------------------------------------------
// x_residual = x + h + mem_context
// ---------------------------------------------------------------------------
__global__ void add3_kernel(const float* __restrict__ a, const float* __restrict__ b,
                            const float* __restrict__ c, float* __restrict__ out,
                            size_t n) {
    size_t i = (size_t)blockIdx.x * blockDim.x + threadIdx.x;
    if (i < n) out[i] = a[i] + b[i] + c[i];
}

// ---------------------------------------------------------------------------
// cumprod along faithful-view axis: gamma flat as [B][H][S][dh], scan over S.
// ---------------------------------------------------------------------------
__global__ void cumprod_kernel(const float* __restrict__ gamma,
                               float* __restrict__ out) {
    const int total = BATCH * NUM_HEADS * HEAD_DIM;
    int idx = blockIdx.x * blockDim.x + threadIdx.x;
    if (idx >= total) return;
    const int d  = idx % HEAD_DIM;
    const int bh = idx / HEAD_DIM;
    size_t base = (size_t)bh * SEQ * HEAD_DIM + d;
    float p = 1.f;
    for (int s = 0; s < SEQ; ++s) {
        p *= gamma[base + (size_t)s * HEAD_DIM];
        out[base + (size_t)s * HEAD_DIM] = p;
    }
}

// ---------------------------------------------------------------------------
// Memory attention: per (b,s,h) block of 128 threads over 64 memory tokens.
// ---------------------------------------------------------------------------
__global__ __launch_bounds__(128)
void attn_kernel(const float* __restrict__ qa, const float* __restrict__ ka,
                 const float* __restrict__ va, __bf16* __restrict__ ctx_bf) {
    const int s = blockIdx.x;
    const int h = blockIdx.y;
    const int b = blockIdx.z;
    const int t = threadIdx.x;
    const int D = MODEL_DIM;

    __shared__ float q_s[HEAD_DIM];
    __shared__ float sc[MEM_TOKENS];

    const size_t qrow = (size_t)(b * SEQ + s) * D + h * HEAD_DIM;
    q_s[t] = qa[qrow + t];
    __syncthreads();

    if (t < MEM_TOKENS) {
        const float* krow = ka + (size_t)(b * MEM_TOKENS + t) * D + h * HEAD_DIM;
        float acc = 0.f;
        for (int d = 0; d < HEAD_DIM; ++d) acc += q_s[d] * krow[d];
        sc[t] = acc * (1.f / sqrtf((float)HEAD_DIM));
    }
    __syncthreads();

    if (t == 0) {
        float mx = sc[0];
        for (int m = 1; m < MEM_TOKENS; ++m) mx = fmaxf(mx, sc[m]);
        float sum = 0.f;
        for (int m = 0; m < MEM_TOKENS; ++m) { float e = __expf(sc[m] - mx); sc[m] = e; sum += e; }
        float inv = 1.f / sum;
        for (int m = 0; m < MEM_TOKENS; ++m) sc[m] *= inv;
    }
    __syncthreads();

    float acc = 0.f;
    for (int m = 0; m < MEM_TOKENS; ++m)
        acc += sc[m] * va[(size_t)(b * MEM_TOKENS + m) * D + h * HEAD_DIM + t];
    ctx_bf[qrow + t] = f2bf(acc);
}

// ---------------------------------------------------------------------------
// Host-side orchestration
// ---------------------------------------------------------------------------
static inline dim3 gemm_grid(int M, int N) { return dim3(N / 128, M / 128); }
static inline dim3 tr_grid(int K, int N)   { return dim3(N / 32,  K / 32); }

extern "C" void kernel_launch(void* const* d_in, const int* in_sizes, int n_in,
                              void* d_out, int out_size, void* d_ws, size_t ws_size,
                              hipStream_t stream) {
    (void)in_sizes; (void)n_in; (void)out_size; (void)ws_size;

    const float* x    = (const float*)d_in[0];
    const float* mem  = (const float*)d_in[1];
    const float* wk   = (const float*)d_in[3];
    const float* wz   = (const float*)d_in[5];
    const float* wg   = (const float*)d_in[6];
    const float* md1  = (const float*)d_in[7];
    const float* md2  = (const float*)d_in[8];
    const float* mcw  = (const float*)d_in[9];
    const float* awq  = (const float*)d_in[10];
    const float* awk  = (const float*)d_in[11];
    const float* awv  = (const float*)d_in[12];
    const float* awo  = (const float*)d_in[13];
    const float* f1   = (const float*)d_in[14];
    const float* f2   = (const float*)d_in[15];
    const float* bk   = (const float*)d_in[17];
    const float* bz   = (const float*)d_in[19];
    const float* bg   = (const float*)d_in[20];
    const float* mb1  = (const float*)d_in[21];
    const float* mb2  = (const float*)d_in[22];
    const float* mcb  = (const float*)d_in[23];
    const float* abq  = (const float*)d_in[24];
    const float* abk  = (const float*)d_in[25];
    const float* abv  = (const float*)d_in[26];
    const float* abo  = (const float*)d_in[27];
    const float* fb1  = (const float*)d_in[28];
    const float* fb2  = (const float*)d_in[29];
    const float* n1w  = (const float*)d_in[30];
    const float* n2w  = (const float*)d_in[31];

    const int D = MODEL_DIM, I = INNER_DIM, F = FFN_DIM;
    const size_t R  = ROWS;          // 4096
    const size_t RM = MROWS;         // 128
    const size_t ND = R * D;         // 8.4M elements

    // ---- bump allocator over workspace ----
    char* ws = (char*)d_ws;
    size_t off = 0;
    auto alloc = [&](size_t bytes) -> void* {
        void* p = ws + off;
        off += (bytes + 255) & ~(size_t)255;
        return p;
    };

    __bf16* wkT  = (__bf16*)alloc((size_t)D * D * 2);
    __bf16* wzT  = (__bf16*)alloc((size_t)D * D * 2);
    __bf16* wgT  = (__bf16*)alloc((size_t)D * D * 2);
    __bf16* mcwT = (__bf16*)alloc((size_t)D * D * 2);
    __bf16* awqT = (__bf16*)alloc((size_t)D * D * 2);
    __bf16* awkT = (__bf16*)alloc((size_t)D * D * 2);
    __bf16* awvT = (__bf16*)alloc((size_t)D * D * 2);
    __bf16* awoT = (__bf16*)alloc((size_t)D * D * 2);
    __bf16* md1T = (__bf16*)alloc((size_t)D * I * 2);
    __bf16* md2T = (__bf16*)alloc((size_t)I * D * 2);
    __bf16* f1T  = (__bf16*)alloc((size_t)D * F * 2);
    __bf16* f2T  = (__bf16*)alloc((size_t)F * D * 2);

    __bf16* x_bf   = (__bf16*)alloc(ND * 2);
    __bf16* xn_bf  = (__bf16*)alloc(ND * 2);
    __bf16* k_bf   = (__bf16*)alloc(ND * 2);
    float*  z_f    = (float*) alloc(ND * 4);
    __bf16* z_bf   = (__bf16*)alloc(ND * 2);
    __bf16* t1_bf  = (__bf16*)alloc(R * I * 2);
    float*  t2_f   = (float*) alloc(ND * 4);
    float*  mc_f   = (float*) alloc(ND * 4);
    float*  gamma  = (float*) alloc(ND * 4);
    float*  gate   = (float*) alloc(ND * 4);
    float*  h_f    = (float*) alloc(ND * 4);
    __bf16* h_bf   = (__bf16*)alloc(ND * 2);
    float*  qa_f   = (float*) alloc(ND * 4);
    __bf16* mem_bf = (__bf16*)alloc(RM * D * 2);
    float*  ka_f   = (float*) alloc(RM * D * 4);
    float*  va_f   = (float*) alloc(RM * D * 4);
    __bf16* ctx_bf = (__bf16*)alloc(ND * 2);
    float*  mctx   = (float*) alloc(ND * 4);
    float*  xr_f   = (float*) alloc(ND * 4);
    __bf16* xn2_bf = (__bf16*)alloc(ND * 2);
    __bf16* t3_bf  = (__bf16*)alloc(R * F * 2);

    float* out_main = (float*)d_out;             // output  (B*S*D)
    float* out_cum  = out_main + ND;             // cumprod (B*S*D)

    const dim3 tb(32, 8);
    const int EW = 256;
    auto eb = [](size_t n) { return (unsigned)((n + 255) / 256); };

    // 1. weight transposes -> bf16 (N x K layout)
    hipLaunchKernelGGL(transpose_to_bf16, tr_grid(D, D), tb, 0, stream, wk,  wkT,  D, D);
    hipLaunchKernelGGL(transpose_to_bf16, tr_grid(D, D), tb, 0, stream, wz,  wzT,  D, D);
    hipLaunchKernelGGL(transpose_to_bf16, tr_grid(D, D), tb, 0, stream, wg,  wgT,  D, D);
    hipLaunchKernelGGL(transpose_to_bf16, tr_grid(D, D), tb, 0, stream, mcw, mcwT, D, D);
    hipLaunchKernelGGL(transpose_to_bf16, tr_grid(D, D), tb, 0, stream, awq, awqT, D, D);
    hipLaunchKernelGGL(transpose_to_bf16, tr_grid(D, D), tb, 0, stream, awk, awkT, D, D);
    hipLaunchKernelGGL(transpose_to_bf16, tr_grid(D, D), tb, 0, stream, awv, awvT, D, D);
    hipLaunchKernelGGL(transpose_to_bf16, tr_grid(D, D), tb, 0, stream, awo, awoT, D, D);
    hipLaunchKernelGGL(transpose_to_bf16, tr_grid(D, I), tb, 0, stream, md1, md1T, D, I);
    hipLaunchKernelGGL(transpose_to_bf16, tr_grid(I, D), tb, 0, stream, md2, md2T, I, D);
    hipLaunchKernelGGL(transpose_to_bf16, tr_grid(D, F), tb, 0, stream, f1,  f1T,  D, F);
    hipLaunchKernelGGL(transpose_to_bf16, tr_grid(F, D), tb, 0, stream, f2,  f2T,  F, D);

    // 2. activations to bf16
    hipLaunchKernelGGL(convert_bf16, dim3(eb(ND)), dim3(EW), 0, stream, x, x_bf, ND);
    hipLaunchKernelGGL(convert_bf16, dim3(eb(RM * D)), dim3(EW), 0, stream, mem, mem_bf, RM * D);

    // 3. xn = rmsnorm(x, n1w)
    hipLaunchKernelGGL(rmsnorm_bf16, dim3((unsigned)R), dim3(EW), 0, stream, x, n1w, xn_bf, D);

    // 4. k_ = xn @ wk + bk        (bf16 out, feeds mcw)
    hipLaunchKernelGGL((gemm_bf16_wmma<0, 2>), gemm_grid(R, D), dim3(256), 0, stream,
                       xn_bf, wkT, bk, (const float*)nullptr, (float*)nullptr, k_bf, (int)R, D, D);
    // 5. z = xn @ wz + bz         (fp32 + bf16)
    hipLaunchKernelGGL((gemm_bf16_wmma<0, 3>), gemm_grid(R, D), dim3(256), 0, stream,
                       xn_bf, wzT, bz, (const float*)nullptr, z_f, z_bf, (int)R, D, D);
    // 6. t1 = gelu(z @ md1 + mb1) (bf16)
    hipLaunchKernelGGL((gemm_bf16_wmma<1, 2>), gemm_grid(R, I), dim3(256), 0, stream,
                       z_bf, md1T, mb1, (const float*)nullptr, (float*)nullptr, t1_bf, (int)R, I, D);
    // 7. t2 = t1 @ md2 + mb2      (fp32)
    hipLaunchKernelGGL((gemm_bf16_wmma<0, 1>), gemm_grid(R, D), dim3(256), 0, stream,
                       t1_bf, md2T, mb2, (const float*)nullptr, t2_f, (__bf16*)nullptr, (int)R, D, I);
    // 8. mc = k_ @ mcw + mcb      (fp32)
    hipLaunchKernelGGL((gemm_bf16_wmma<0, 1>), gemm_grid(R, D), dim3(256), 0, stream,
                       k_bf, mcwT, mcb, (const float*)nullptr, mc_f, (__bf16*)nullptr, (int)R, D, D);
    // 9. gamma = sigmoid(t2 + mc)
    hipLaunchKernelGGL(sigmoid_add_kernel, dim3(eb(ND)), dim3(EW), 0, stream, t2_f, mc_f, gamma, ND);
    // 10. gate = sigmoid(x @ wg + bg)
    hipLaunchKernelGGL((gemm_bf16_wmma<2, 1>), gemm_grid(R, D), dim3(256), 0, stream,
                       x_bf, wgT, bg, (const float*)nullptr, gate, (__bf16*)nullptr, (int)R, D, D);
    // 11. h_t = z*gate + gamma*x
    hipLaunchKernelGGL(ht_kernel, dim3(eb(ND)), dim3(EW), 0, stream, z_f, gate, gamma, x, h_f, h_bf, ND);
    // 12. cumprod -> second output
    hipLaunchKernelGGL(cumprod_kernel, dim3(16), dim3(EW), 0, stream, gamma, out_cum);
    // 13. qa = h_t @ awq + abq
    hipLaunchKernelGGL((gemm_bf16_wmma<0, 1>), gemm_grid(R, D), dim3(256), 0, stream,
                       h_bf, awqT, abq, (const float*)nullptr, qa_f, (__bf16*)nullptr, (int)R, D, D);
    // 14. ka / va from memory bank (M = 128 rows)
    hipLaunchKernelGGL((gemm_bf16_wmma<0, 1>), gemm_grid(RM, D), dim3(256), 0, stream,
                       mem_bf, awkT, abk, (const float*)nullptr, ka_f, (__bf16*)nullptr, (int)RM, D, D);
    hipLaunchKernelGGL((gemm_bf16_wmma<0, 1>), gemm_grid(RM, D), dim3(256), 0, stream,
                       mem_bf, awvT, abv, (const float*)nullptr, va_f, (__bf16*)nullptr, (int)RM, D, D);
    // 15. attention over memory tokens -> ctx (bf16)
    hipLaunchKernelGGL(attn_kernel, dim3(SEQ, NUM_HEADS, BATCH), dim3(HEAD_DIM), 0, stream,
                       qa_f, ka_f, va_f, ctx_bf);
    // 16. mem_context = ctx @ awo + abo
    hipLaunchKernelGGL((gemm_bf16_wmma<0, 1>), gemm_grid(R, D), dim3(256), 0, stream,
                       ctx_bf, awoT, abo, (const float*)nullptr, mctx, (__bf16*)nullptr, (int)R, D, D);
    // 17. x_residual = x + h_t + mem_context
    hipLaunchKernelGGL(add3_kernel, dim3(eb(ND)), dim3(EW), 0, stream, x, h_f, mctx, xr_f, ND);
    // 18. xn2 = rmsnorm(x_residual, n2w)
    hipLaunchKernelGGL(rmsnorm_bf16, dim3((unsigned)R), dim3(EW), 0, stream, xr_f, n2w, xn2_bf, D);
    // 19. t3 = gelu(xn2 @ f1 + fb1)
    hipLaunchKernelGGL((gemm_bf16_wmma<1, 2>), gemm_grid(R, F), dim3(256), 0, stream,
                       xn2_bf, f1T, fb1, (const float*)nullptr, (float*)nullptr, t3_bf, (int)R, F, D);
    // 20. output = x_residual + (t3 @ f2 + fb2)   -> first output
    hipLaunchKernelGGL((gemm_bf16_wmma<0, 1>), gemm_grid(R, D), dim3(256), 0, stream,
                       t3_bf, f2T, fb2, xr_f, out_main, (__bf16*)nullptr, (int)R, D, F);
}
